// MOE_62500364091814
// MI455X (gfx1250) — compile-verified, matrix operand order checked by reading, not run on hardware
//
#include <hip/hip_runtime.h>
#include <hip/hip_bf16.h>
#include <math.h>

// Problem dims (fixed by the reference)
#define BB 8192
#define DD 1024
#define HH 4096
#define EE 8
#define PP (2*BB)          // total (token, expert) pairs = top-2 routing
#define MT_MAX 520         // sum ceil(Ne/32) <= 2B/32 + E = 520
#define ROWSTRIDE 40       // LDS A-tile row stride in bf16 elements (80B, bank-spread)

typedef __bf16 bhalf;
typedef bhalf        v16bf __attribute__((ext_vector_type(16)));
typedef float        v8f   __attribute__((ext_vector_type(8)));
typedef float        v4f   __attribute__((ext_vector_type(4)));
typedef unsigned int v4u   __attribute__((ext_vector_type(4)));

union AF { v4u q[2]; v16bf v; };   // 32-byte WMMA 16-bit operand fragment

// ---------------------------------------------------------------- meta zero
__global__ void k_zero_meta(int* meta) {
    if (threadIdx.x < 64) meta[threadIdx.x] = 0;
}

// ---------------------------------------------------------------- fp32 -> bf16
__global__ void k_cvt_bf16(const float* __restrict__ s, bhalf* __restrict__ d, int n4) {
    int i = blockIdx.x * blockDim.x + threadIdx.x;
    if (i < n4) {
        v4f v = *(const v4f*)(s + (size_t)i * 4);
        size_t o = (size_t)i * 4;
        #pragma unroll
        for (int j = 0; j < 4; j++) d[o + j] = (bhalf)v[j];
    }
}

// ---------------------------------------------------------------- gating: one wave per token
__global__ void k_gate(const float* __restrict__ inp, const float* __restrict__ gw,
                       const float* __restrict__ gb, int* __restrict__ tok_e,
                       float* __restrict__ tok_g, int* __restrict__ counts) {
    int wave = threadIdx.x >> 5, lane = threadIdx.x & 31;
    int b = blockIdx.x * 8 + wave;           // 1024 blocks * 8 waves = 8192 tokens
    const float* x = inp + (size_t)b * DD;
    float lg[EE];
    #pragma unroll
    for (int e = 0; e < EE; e++) {
        const float* w = gw + (size_t)e * DD;
        float a = 0.f;
        for (int d = lane; d < DD; d += 32) a += x[d] * w[d];
        #pragma unroll
        for (int m = 16; m > 0; m >>= 1) a += __shfl_xor(a, m, 32);
        lg[e] = a + gb[e];
    }
    if (lane == 0) {
        int i0 = 0; float v0 = lg[0];
        #pragma unroll
        for (int e = 1; e < EE; e++) if (lg[e] > v0) { v0 = lg[e]; i0 = e; }
        int i1 = (i0 == 0) ? 1 : 0; float v1 = lg[i1];
        #pragma unroll
        for (int e = 0; e < EE; e++) if (e != i0 && lg[e] > v1) { v1 = lg[e]; i1 = e; }
        // softmax over the 2 top values (v0 >= v1)
        float e1 = __expf(v1 - v0);
        float s  = 1.f + e1;
        tok_e[b * 2 + 0] = i0;  tok_e[b * 2 + 1] = i1;
        tok_g[b * 2 + 0] = 1.f / s;  tok_g[b * 2 + 1] = e1 / s;
        atomicAdd(&counts[i0], 1);
        atomicAdd(&counts[i1], 1);
    }
}

// ---------------------------------------------------------------- scan (E=8, trivial)
// meta layout: [0..7]=counts [8..16]=row offsets [17..24]=cursors [25..33]=tile offsets
__global__ void k_scan(int* meta) {
    int* counts = meta; int* off = meta + 8; int* cur = meta + 17; int* to = meta + 25;
    int o = 0, t = 0;
    for (int e = 0; e < EE; e++) {
        off[e] = o; cur[e] = o; to[e] = t;
        o += counts[e];
        t += (counts[e] + 31) >> 5;          // 32-row M tiles
    }
    off[EE] = o; to[EE] = t;
}

// ---------------------------------------------------------------- scatter rows per expert
__global__ void k_scatter(const int* __restrict__ tok_e, const float* __restrict__ tok_g,
                          int* __restrict__ meta, int* __restrict__ slot_token,
                          float* __restrict__ slot_gate, int* __restrict__ pair_pos) {
    int idx = blockIdx.x * blockDim.x + threadIdx.x;
    if (idx < PP) {
        int e   = tok_e[idx];
        int pos = atomicAdd(&meta[17 + e], 1);
        slot_token[pos] = idx >> 1;
        slot_gate[pos]  = tok_g[idx];
        pair_pos[idx]   = pos;
    }
}

// ---------------------------------------------------------------- GEMM main-loop helper
// Stage a 32x32 bf16 A tile (rows given by s_row[]) into LDS via async load,
// then each wave does 8 WMMAs (32x64 wave tile) against 4 B fragments.
__device__ __forceinline__ void gemm_step(
    bhalf* As, const bhalf* __restrict__ X, const int* s_row, size_t K,
    const bhalf* __restrict__ Wb, int nb, int k0,
    int tid, int col, int hi, v8f acc[8]) {
    __syncthreads();                          // protect previous iteration's LDS reads
    {
        int row = tid >> 3, ch = tid & 7;     // 32 rows x 8 chunks of 8B
        unsigned lds = (unsigned)(size_t)&As[row * ROWSTRIDE + ch * 4];
        const bhalf* g = X + (size_t)s_row[row] * K + k0 + ch * 4;
        asm volatile("global_load_async_to_lds_b64 %0, %1, off"
                     :: "v"(lds), "v"(g) : "memory");
    }
    asm volatile("s_wait_asynccnt 0x0" ::: "memory");
    __syncthreads();
    AF a0, a1;                                // ISA 16-bit A layout: K {0..7,16..23}/{8..15,24..31}
    a0.q[0] = *(const v4u*)(&As[col * ROWSTRIDE + hi * 8]);
    a0.q[1] = *(const v4u*)(&As[col * ROWSTRIDE + hi * 8 + 16]);
    a1.q[0] = *(const v4u*)(&As[(16 + col) * ROWSTRIDE + hi * 8]);
    a1.q[1] = *(const v4u*)(&As[(16 + col) * ROWSTRIDE + hi * 8 + 16]);
    #pragma unroll
    for (int j = 0; j < 4; j++) {
        const bhalf* bp = Wb + (size_t)(nb + j * 16 + col) * K + k0 + hi * 8;
        AF bb;
        bb.q[0] = *(const v4u*)bp;
        bb.q[1] = *(const v4u*)(bp + 16);
        __builtin_prefetch((const void*)(bp + 32), 0, 3);   // next K step
        acc[j]     = __builtin_amdgcn_wmma_f32_16x16x32_bf16(
            false, a0.v, false, bb.v, (short)0, acc[j], false, false);
        acc[4 + j] = __builtin_amdgcn_wmma_f32_16x16x32_bf16(
            false, a1.v, false, bb.v, (short)0, acc[4 + j], false, false);
    }
}

// ---------------------------------------------------------------- GEMM1: h = relu(x @ W1^T + b1), bf16 out
// grid (MT_MAX, 8), block 256 (8 waves). Block tile 32x512, wave tile 32x64, K step 32.
__global__ __launch_bounds__(256) void k_gemm1(
    const bhalf* __restrict__ X, const bhalf* __restrict__ W,
    const float* __restrict__ bias, const int* __restrict__ meta,
    const int* __restrict__ slot_token, bhalf* __restrict__ Hb) {
    const int* off = meta + 8; const int* to = meta + 25;
    int mt = blockIdx.x;
    if (mt >= to[EE]) return;
    int e = 0;
    #pragma unroll
    for (int i = 1; i < EE; i++) if (mt >= to[i]) e = i;
    int lt    = mt - to[e];
    int slot0 = off[e] + lt * 32;
    int valid = off[e + 1] - slot0; if (valid > 32) valid = 32;

    __shared__ __align__(16) bhalf As[32 * ROWSTRIDE];
    __shared__ int s_row[32];
    int tid = threadIdx.x;
    if (tid < 32) s_row[tid] = slot_token[slot0 + (tid < valid ? tid : 0)];

    int wave = tid >> 5, lane = tid & 31;
    int col = lane & 15, hi = lane >> 4;
    int nb = blockIdx.y * 512 + wave * 64;
    const bhalf* Wb = W + (size_t)e * HH * DD;

    v8f acc[8];
    #pragma unroll
    for (int j = 0; j < 8; j++)
        #pragma unroll
        for (int i = 0; i < 8; i++) acc[j][i] = 0.f;

    for (int k0 = 0; k0 < DD; k0 += 32)
        gemm_step(As, X, s_row, DD, Wb, nb, k0, tid, col, hi, acc);

    #pragma unroll
    for (int mh = 0; mh < 2; mh++)
        #pragma unroll
        for (int j = 0; j < 4; j++) {
            int n = nb + j * 16 + col;
            float bv = bias[(size_t)e * HH + n];
            #pragma unroll
            for (int i = 0; i < 8; i++) {
                int m = mh * 16 + hi * 8 + i;
                if (m < valid) {
                    float v = acc[mh * 4 + j][i] + bv;
                    v = v > 0.f ? v : 0.f;
                    Hb[(size_t)(slot0 + m) * HH + n] = (bhalf)v;
                }
            }
        }
}

// ---------------------------------------------------------------- GEMM2: c = gate*exp(relu(h @ W2^T + b2))
// grid (MT_MAX, 2), block 256.
__global__ __launch_bounds__(256) void k_gemm2(
    const bhalf* __restrict__ Hb, const bhalf* __restrict__ W,
    const float* __restrict__ bias, const int* __restrict__ meta,
    const float* __restrict__ slot_gate, float* __restrict__ contrib) {
    const int* off = meta + 8; const int* to = meta + 25;
    int mt = blockIdx.x;
    if (mt >= to[EE]) return;
    int e = 0;
    #pragma unroll
    for (int i = 1; i < EE; i++) if (mt >= to[i]) e = i;
    int lt    = mt - to[e];
    int slot0 = off[e] + lt * 32;
    int valid = off[e + 1] - slot0; if (valid > 32) valid = 32;

    __shared__ __align__(16) bhalf As[32 * ROWSTRIDE];
    __shared__ int s_row[32];
    __shared__ float s_gate[32];
    int tid = threadIdx.x;
    if (tid < 32) {
        int r = slot0 + tid; if (r > PP - 1) r = PP - 1;
        s_row[tid]  = r;                       // h rows are slot-contiguous, no gather
        s_gate[tid] = (tid < valid) ? slot_gate[slot0 + tid] : 0.f;
    }

    int wave = tid >> 5, lane = tid & 31;
    int col = lane & 15, hi = lane >> 4;
    int nb = blockIdx.y * 512 + wave * 64;
    const bhalf* Wb = W + (size_t)e * DD * HH;

    v8f acc[8];
    #pragma unroll
    for (int j = 0; j < 8; j++)
        #pragma unroll
        for (int i = 0; i < 8; i++) acc[j][i] = 0.f;

    for (int k0 = 0; k0 < HH; k0 += 32)
        gemm_step(As, Hb, s_row, HH, Wb, nb, k0, tid, col, hi, acc);

    #pragma unroll
    for (int mh = 0; mh < 2; mh++)
        #pragma unroll
        for (int j = 0; j < 4; j++) {
            int n = nb + j * 16 + col;
            float bv = bias[(size_t)e * DD + n];
            #pragma unroll
            for (int i = 0; i < 8; i++) {
                int m = mh * 16 + hi * 8 + i;
                if (m < valid) {
                    float v = acc[mh * 4 + j][i] + bv;
                    v = v > 0.f ? v : 0.f;
                    contrib[(size_t)(slot0 + m) * DD + n] = s_gate[m] * __expf(v);
                }
            }
        }
}

// ---------------------------------------------------------------- combine: log(c0 + c1 + eps)
__global__ void k_combine(const float* __restrict__ contrib, const int* __restrict__ pair_pos,
                          float* __restrict__ out) {
    int i4 = blockIdx.x * blockDim.x + threadIdx.x;
    size_t i = (size_t)i4 * 4;
    if (i < (size_t)BB * DD) {
        int b = (int)(i >> 10);         // D = 1024
        int d = (int)(i & 1023);
        int p0 = pair_pos[b * 2], p1 = pair_pos[b * 2 + 1];
        v4f a = *(const v4f*)(contrib + (size_t)p0 * DD + d);
        v4f c = *(const v4f*)(contrib + (size_t)p1 * DD + d);
        v4f o;
        #pragma unroll
        for (int j = 0; j < 4; j++)
            o[j] = logf(a[j] + c[j] + 2.220446049250313e-16f);
        *(v4f*)(out + i) = o;
    }
}

// ----------------------------------------------------------------
extern "C" void kernel_launch(void* const* d_in, const int* in_sizes, int n_in,
                              void* d_out, int out_size, void* d_ws, size_t ws_size,
                              hipStream_t stream) {
    const float* inp = (const float*)d_in[0];
    const float* gw  = (const float*)d_in[1];
    const float* gb  = (const float*)d_in[2];
    const float* W1  = (const float*)d_in[3];
    const float* b1  = (const float*)d_in[4];
    const float* W2  = (const float*)d_in[5];
    const float* b2  = (const float*)d_in[6];
    float* out = (float*)d_out;

    char* ws = (char*)d_ws;
    size_t o = 0;
    bhalf* inp_b = (bhalf*)(ws + o); o += (size_t)BB * DD * 2;
    bhalf* w1_b  = (bhalf*)(ws + o); o += (size_t)EE * HH * DD * 2;
    bhalf* w2_b  = (bhalf*)(ws + o); o += (size_t)EE * DD * HH * 2;
    bhalf* h_b   = (bhalf*)(ws + o); o += (size_t)PP * HH * 2;
    float* con   = (float*)(ws + o); o += (size_t)PP * DD * 4;
    int*   tok_e = (int*)(ws + o);   o += (size_t)PP * 4;
    float* tok_g = (float*)(ws + o); o += (size_t)PP * 4;
    int*   s_tok = (int*)(ws + o);   o += (size_t)PP * 4;
    float* s_gt  = (float*)(ws + o); o += (size_t)PP * 4;
    int*   p_pos = (int*)(ws + o);   o += (size_t)PP * 4;
    int*   meta  = (int*)(ws + o);   o += 256;

    k_zero_meta<<<1, 64, 0, stream>>>(meta);

    int n4_inp = (BB * DD) / 4;           // 2,097,152
    int n4_w   = (EE * HH * DD) / 4;      // 8,388,608
    k_cvt_bf16<<<(n4_inp + 255) / 256, 256, 0, stream>>>(inp, inp_b, n4_inp);
    k_cvt_bf16<<<(n4_w   + 255) / 256, 256, 0, stream>>>(W1, w1_b, n4_w);
    k_cvt_bf16<<<(n4_w   + 255) / 256, 256, 0, stream>>>(W2, w2_b, n4_w);

    k_gate<<<BB / 8, 256, 0, stream>>>(inp, gw, gb, tok_e, tok_g, meta);
    k_scan<<<1, 1, 0, stream>>>(meta);
    k_scatter<<<PP / 256, 256, 0, stream>>>(tok_e, tok_g, meta, s_tok, s_gt, p_pos);

    k_gemm1<<<dim3(MT_MAX, HH / 512), 256, 0, stream>>>(inp_b, w1_b, b1, meta, s_tok, h_b);
    k_gemm2<<<dim3(MT_MAX, DD / 512), 256, 0, stream>>>(h_b, w2_b, b2, meta, s_gt, con);

    k_combine<<<(BB * DD / 4) / 256, 256, 0, stream>>>(con, p_pos, out);
}
